// jointLayer_9689446220236
// MI455X (gfx1250) — matrix operand order repaired and checked by reading, not compiled, not run on hardware
//
#include <hip/hip_runtime.h>
#include <hip/hip_bf16.h>

// ---------------------------------------------------------------------------
// Non-local block for MI455X (gfx1250): bf16 WMMA GEMMs, fp32 accumulate.
// All heavy math runs through v_wmma_f32_16x16x32_bf16.
// B operands are kept column-major (N x K row-major) in global memory so both
// A and B tile staging are contiguous b128 copies (no scalar LDS transpose).
// ---------------------------------------------------------------------------

typedef __attribute__((ext_vector_type(16))) __bf16 v16bf;
typedef __attribute__((ext_vector_type(8)))  __bf16 v8bf;
typedef __attribute__((ext_vector_type(8)))  float  v8f;

struct BF16x16 { union { v8bf h[2]; v16bf v; }; };

constexpr int BATCH = 32;
constexpr int CH    = 1536;
constexpr int CID   = 768;     // inter channels
constexpr int HJ    = 28;
constexpr int NPOS  = 784;     // 28*28 query positions
constexpr int NPOOL = 196;     // 14*14 pooled positions
constexpr int NPAD  = 224;     // pooled dim padded to multiple of 32
constexpr int O3    = 2304;    // CID*3 fused theta|phi|g output channels
constexpr int NW    = CID * CH; // 1179648: elements in each weight matrix

// ---------------------------------------------------------------------------
// Pack x (B*4, C, 14, 14) fp32 -> joint Xj (B, 784, 1536) bf16 row-major.
// ---------------------------------------------------------------------------
__global__ void pack_x_kernel(const float* __restrict__ x, __bf16* __restrict__ Xj)
{
    size_t idx = (size_t)blockIdx.x * blockDim.x + threadIdx.x;
    if (idx >= (size_t)BATCH * NPOS * CH) return;
    int c = (int)(idx % CH);
    size_t tt = idx / CH;
    int n = (int)(tt % NPOS);
    int b = (int)(tt / NPOS);
    int h = n / HJ, w = n % HJ;
    int v  = ((h >= 14) ? 2 : 0) + ((w >= 14) ? 1 : 0);
    int hw = ((h >= 14) ? h - 14 : h) * 14 + ((w >= 14) ? w - 14 : w);
    Xj[idx] = (__bf16)x[((size_t)(b * 4 + v) * CH + c) * 196 + hw];
}

// ---------------------------------------------------------------------------
// Weights are already (out_ch x in_ch) row-major == the N x K column-major
// B layout the GEMM wants. Pack = pure bf16 cast of 4 concatenated arrays:
//   wB[0 .. NW)      = w_theta (768x1536)
//   wB[NW .. 2NW)    = w_phi
//   wB[2NW .. 3NW)   = w_g       -> together: wT3' (2304 x 1536)
//   wB[3NW .. 4NW)   = w_W (1536x768) -> wTW' (1536 x 768)
// ---------------------------------------------------------------------------
__global__ void pack_w_kernel(const float* __restrict__ wt, const float* __restrict__ wp,
                              const float* __restrict__ wg, const float* __restrict__ wW,
                              __bf16* __restrict__ wB)
{
    int idx = blockIdx.x * blockDim.x + threadIdx.x;
    if (idx >= 4 * NW) return;
    float s = (idx < NW)     ? wt[idx]
            : (idx < 2 * NW) ? wp[idx - NW]
            : (idx < 3 * NW) ? wg[idx - 2 * NW]
            :                  wW[idx - 3 * NW];
    wB[idx] = (__bf16)s;
}

// ---------------------------------------------------------------------------
// Generic bf16 GEMM: C[b](MxN) = A[b](MxK) * Bt[b]^T,  Bt stored (N x K).
// Block: 256 threads = 8 waves, block tile 64(M) x 256(N), K step 32.
// Wave (wr,wc) owns 32(M) x 64(N) -> 2x4 grid of 16x16 WMMA tiles
// (8 v_wmma per K-step per wave). Global->reg loads for step k+32 are issued
// before the WMMAs of step k (software pipeline).
// LDS: both tiles k-contiguous with padded stride 40 bf16; fragment loads are
// two contiguous 16B ds reads matching CDNA5 16-bit A(16x32)/B(32x16) layouts.
// mode 0: bf16 out + segmented bias (theta|phi|g)
// mode 1: fp32 out (attention logits f)
// mode 2: bf16 out (y)
// mode 3: fp32 out with bias + BN + residual, written to (B,C,28,28)
// ---------------------------------------------------------------------------
__global__ __launch_bounds__(256) void gemm_bf16_kernel(
    const __bf16* __restrict__ A,  long long ldA, long long strideA,
    const __bf16* __restrict__ Bt, long long ldB, long long strideB,
    int M, int N, int K, int mode,
    void* __restrict__ Cout, long long ldC, long long strideC,
    const float* __restrict__ p0, const float* __restrict__ p1,
    const float* __restrict__ p2, const float* __restrict__ p3,
    const float* __restrict__ p4, const float* __restrict__ xsrc)
{
    __shared__ __bf16 ldsA[64 * 40];    // 64 A-rows x 32 k, padded stride 40
    __shared__ __bf16 ldsB[256 * 40];   // 256 B-cols(n) x 32 k, stride 40

    const int t    = threadIdx.x;
    const int lane = t & 31;
    const int wave = t >> 5;
    const int wr   = wave >> 2;   // 0..1 : 32-row strip
    const int wc   = wave & 3;    // 0..3 : 64-col strip
    const int bM   = blockIdx.y * 64;
    const int bN   = blockIdx.x * 256;
    const int b    = blockIdx.z;

    const __bf16* Ap = A  + (size_t)b * strideA;
    const __bf16* Bp = Bt + (size_t)b * strideB;

    v8f acc[2][4] = {};

    const int prow = t >> 2;         // 0..63
    const int pkq  = (t & 3) * 8;    // 0,8,16,24
    const int gr   = bM + prow;      // A row for staging

    auto loadA_g = [&](int k0) -> v8bf {
        v8bf v = {};
        if (gr < M) v = *(const v8bf*)(Ap + (size_t)gr * ldA + k0 + pkq);
        return v;
    };
    auto loadB_g = [&](int k0, int q) -> v8bf {
        v8bf v = {};
        int gn = bN + prow + 64 * q;
        if (gn < N) v = *(const v8bf*)(Bp + (size_t)gn * ldB + k0 + pkq);
        return v;
    };

    // pipeline prologue
    v8bf areg  = loadA_g(0);
    v8bf breg0 = loadB_g(0, 0), breg1 = loadB_g(0, 1);
    v8bf breg2 = loadB_g(0, 2), breg3 = loadB_g(0, 3);

    for (int k0 = 0; k0 < K; k0 += 32) {
        __syncthreads();                       // LDS free to overwrite
        *(v8bf*)&ldsA[prow * 40 + pkq]         = areg;
        *(v8bf*)&ldsB[(prow      ) * 40 + pkq] = breg0;
        *(v8bf*)&ldsB[(prow +  64) * 40 + pkq] = breg1;
        *(v8bf*)&ldsB[(prow + 128) * 40 + pkq] = breg2;
        *(v8bf*)&ldsB[(prow + 192) * 40 + pkq] = breg3;
        __syncthreads();                       // tiles visible

        const int k1 = k0 + 32;
        if (k1 < K) {                          // prefetch next step into regs
            areg  = loadA_g(k1);
            breg0 = loadB_g(k1, 0); breg1 = loadB_g(k1, 1);
            breg2 = loadB_g(k1, 2); breg3 = loadB_g(k1, 3);
            if (gr < M && k1 + 32 < K)         // L2 prefetch one step further
                __builtin_prefetch(Ap + (size_t)gr * ldA + k1 + 32 + pkq, 0, 3);
        }

        // ---- fragment loads ----
        // A frag: m = lane&15, kh = lane>>4; elems 0..7 <- k=kh*8+0..7,
        //         elems 8..15 <- k=16+kh*8+0..7
        // B frag (from ldsB[n][k]): n = lane&15, kh = lane>>4;
        //         elems 0..7 <- k=kh*8+0..7, elems 8..15 <- k=16+kh*8+0..7
        const int fm = lane & 15;
        const int kh = lane >> 4;
        BF16x16 af[2], bfr[4];
#pragma unroll
        for (int i = 0; i < 2; ++i) {
            const __bf16* p = &ldsA[(wr * 32 + i * 16 + fm) * 40 + kh * 8];
            af[i].h[0] = *(const v8bf*)p;
            af[i].h[1] = *(const v8bf*)(p + 16);
        }
#pragma unroll
        for (int j = 0; j < 4; ++j) {
            const __bf16* p = &ldsB[(wc * 64 + j * 16 + fm) * 40 + kh * 8];
            bfr[j].h[0] = *(const v8bf*)p;
            bfr[j].h[1] = *(const v8bf*)(p + 16);
        }
#pragma unroll
        for (int i = 0; i < 2; ++i)
#pragma unroll
            for (int j = 0; j < 4; ++j)
                acc[i][j] = __builtin_amdgcn_wmma_f32_16x16x32_bf16(
                    false, af[i].v, false, bfr[j].v, (short)0, acc[i][j], false, false);
    }

    // ---- epilogue. C/D layout: lane&15 = column, (lane>>4)*8 + r = row ----
    const int cn  = lane & 15;
    const int chh = lane >> 4;
#pragma unroll
    for (int i = 0; i < 2; ++i) {
#pragma unroll
        for (int j = 0; j < 4; ++j) {
            int gn = bN + wc * 64 + j * 16 + cn;
            if (gn >= N) continue;
            int gm0 = bM + wr * 32 + i * 16 + chh * 8;
#pragma unroll
            for (int r = 0; r < 8; ++r) {
                int gm = gm0 + r;
                if (gm >= M) continue;
                float v = acc[i][j][r];
                if (mode == 0) {
                    float bias = (gn < CID) ? p0[gn]
                               : (gn < 2 * CID) ? p1[gn - CID] : p2[gn - 2 * CID];
                    ((__bf16*)Cout)[(size_t)b * strideC + (size_t)gm * ldC + gn] = (__bf16)(v + bias);
                } else if (mode == 1) {
                    ((float*)Cout)[(size_t)b * strideC + (size_t)gm * ldC + gn] = v;
                } else if (mode == 2) {
                    ((__bf16*)Cout)[(size_t)b * strideC + (size_t)gm * ldC + gn] = (__bf16)v;
                } else {
                    // mode 3: bias + batchnorm + residual, out (B, C, 28, 28)
                    float val = v + p0[gn];
                    float inv = rsqrtf(p4[gn] + 1e-5f);
                    val = (val - p3[gn]) * (p1[gn] * inv) + p2[gn];
                    int h = gm / HJ, w = gm % HJ;
                    int vi = ((h >= 14) ? 2 : 0) + ((w >= 14) ? 1 : 0);
                    int hw = ((h >= 14) ? h - 14 : h) * 14 + ((w >= 14) ? w - 14 : w);
                    float resid = xsrc[((size_t)(b * 4 + vi) * CH + gn) * 196 + hw];
                    ((float*)Cout)[((size_t)b * CH + gn) * NPOS + gm] = val + resid;
                }
            }
        }
    }
}

// ---------------------------------------------------------------------------
// 2x2 max-pool of phi/g channels from fused TPG, with layout transforms into
// the GEMM's column-major B layout:
//  phiP (B, 224, 768): position-major -> B' for f = theta * phi^T (N=224,K=768)
//  gT   (B, 768, 224): channel-major  -> B' for y = attn * g      (N=768,K=224)
// Pooled positions 196..223 are zero padding (softmax masks them to 0).
// ---------------------------------------------------------------------------
__global__ void pool_kernel(const __bf16* __restrict__ TPG,
                            __bf16* __restrict__ phiP, __bf16* __restrict__ gT)
{
    int idx = blockIdx.x * blockDim.x + threadIdx.x;
    if (idx >= BATCH * CID * NPAD) return;
    int mp = idx % NPAD;
    int r  = idx / NPAD;
    int o  = r % CID;
    int b  = r / CID;
    float pv = 0.f, gv = 0.f;
    if (mp < NPOOL) {
        int ph = mp / 14, pw = mp % 14;
        float pm = -3.4e38f, gm = -3.4e38f;
#pragma unroll
        for (int dh = 0; dh < 2; ++dh)
#pragma unroll
            for (int dw = 0; dw < 2; ++dw) {
                int n = (2 * ph + dh) * HJ + (2 * pw + dw);
                const __bf16* row = TPG + ((size_t)b * NPOS + n) * O3;
                pm = fmaxf(pm, (float)row[CID + o]);
                gm = fmaxf(gm, (float)row[2 * CID + o]);
            }
        pv = pm; gv = gm;
    }
    phiP[((size_t)b * NPAD + mp) * CID + o] = (__bf16)pv;
    gT[((size_t)b * CID + o) * NPAD + mp]   = (__bf16)gv;
}

// ---------------------------------------------------------------------------
// Row softmax over 196 valid columns (224 padded); one wave32 per row.
// ---------------------------------------------------------------------------
__global__ __launch_bounds__(256) void softmax_kernel(const float* __restrict__ f,
                                                      __bf16* __restrict__ attn)
{
    int row  = blockIdx.x * 8 + (threadIdx.x >> 5);
    int lane = threadIdx.x & 31;
    if (row >= BATCH * NPOS) return;
    const float* fr = f + (size_t)row * NPAD;
    float v[7];
    float mx = -3.4e38f;
#pragma unroll
    for (int i = 0; i < 7; ++i) {
        int c = i * 32 + lane;
        v[i] = (c < NPOOL) ? fr[c] : -3.4e38f;
        mx = fmaxf(mx, v[i]);
    }
#pragma unroll
    for (int off = 16; off > 0; off >>= 1) mx = fmaxf(mx, __shfl_xor(mx, off, 32));
    float s = 0.f;
#pragma unroll
    for (int i = 0; i < 7; ++i) {
        int c = i * 32 + lane;
        float e = (c < NPOOL) ? __expf(v[i] - mx) : 0.f;
        v[i] = e; s += e;
    }
#pragma unroll
    for (int off = 16; off > 0; off >>= 1) s += __shfl_xor(s, off, 32);
    float inv = 1.f / s;
    __bf16* ar = attn + (size_t)row * NPAD;
#pragma unroll
    for (int i = 0; i < 7; ++i) {
        int c = i * 32 + lane;                 // always < 224
        ar[c] = (__bf16)(v[i] * inv);          // padded cols get exactly 0
    }
}

// ---------------------------------------------------------------------------
extern "C" void kernel_launch(void* const* d_in, const int* in_sizes, int n_in,
                              void* d_out, int out_size, void* d_ws, size_t ws_size,
                              hipStream_t stream)
{
    (void)in_sizes; (void)n_in; (void)out_size; (void)ws_size;
    const float* x       = (const float*)d_in[0];
    const float* w_theta = (const float*)d_in[1];
    const float* b_theta = (const float*)d_in[2];
    const float* w_phi   = (const float*)d_in[3];
    const float* b_phi   = (const float*)d_in[4];
    const float* w_g     = (const float*)d_in[5];
    const float* b_g     = (const float*)d_in[6];
    const float* w_W     = (const float*)d_in[7];
    const float* b_W     = (const float*)d_in[8];
    const float* gamma   = (const float*)d_in[9];
    const float* beta    = (const float*)d_in[10];
    const float* mean    = (const float*)d_in[11];
    const float* var     = (const float*)d_in[12];

    // workspace carve-out (~296 MB), 256B aligned chunks
    char* ws = (char*)d_ws;
    size_t off = 0;
    auto carve = [&](size_t bytes) {
        void* p = ws + off;
        off = (off + bytes + 255) & ~(size_t)255;
        return p;
    };
    __bf16* Xj   = (__bf16*)carve((size_t)BATCH * NPOS * CH  * 2);
    __bf16* wB   = (__bf16*)carve((size_t)4 * NW * 2);
    __bf16* TPG  = (__bf16*)carve((size_t)BATCH * NPOS * O3  * 2);
    __bf16* phiP = (__bf16*)carve((size_t)BATCH * NPAD * CID * 2);
    __bf16* gT   = (__bf16*)carve((size_t)BATCH * CID * NPAD * 2);
    float*  fbuf = (float*) carve((size_t)BATCH * NPOS * NPAD * 4);
    __bf16* attn = (__bf16*)carve((size_t)BATCH * NPOS * NPAD * 2);
    __bf16* ybuf = (__bf16*)carve((size_t)BATCH * NPOS * CID * 2);
    __bf16* wT3  = wB;               // (2304 x 1536)
    __bf16* wTW  = wB + (size_t)3 * NW; // (1536 x 768)

    // 1) pack inputs & weights to bf16
    {
        size_t tot = (size_t)BATCH * NPOS * CH;
        pack_x_kernel<<<(unsigned)((tot + 255) / 256), 256, 0, stream>>>(x, Xj);
    }
    pack_w_kernel<<<(4 * NW + 255) / 256, 256, 0, stream>>>(w_theta, w_phi, w_g, w_W, wB);
    // 2) fused theta|phi|g projection: (784x1536) x (2304x1536)^T + biases
    gemm_bf16_kernel<<<dim3(O3 / 256, 13, BATCH), 256, 0, stream>>>(
        Xj, CH, (long long)NPOS * CH, wT3, CH, 0,
        NPOS, O3, CH, 0,
        TPG, O3, (long long)NPOS * O3,
        b_theta, b_phi, b_g, nullptr, nullptr, nullptr);
    // 3) max-pool phi/g + layout transform
    {
        int tot = BATCH * CID * NPAD;
        pool_kernel<<<(tot + 255) / 256, 256, 0, stream>>>(TPG, phiP, gT);
    }
    // 4) f = theta * phi^T : (784x768) x (224x768)^T, fp32 logits
    gemm_bf16_kernel<<<dim3(1, 13, BATCH), 256, 0, stream>>>(
        TPG, O3, (long long)NPOS * O3, phiP, CID, (long long)NPAD * CID,
        NPOS, NPAD, CID, 1,
        fbuf, NPAD, (long long)NPOS * NPAD,
        nullptr, nullptr, nullptr, nullptr, nullptr, nullptr);
    // 5) softmax rows -> bf16 attn
    softmax_kernel<<<(BATCH * NPOS) / 8, 256, 0, stream>>>(fbuf, attn);
    // 6) y = attn * g : (784x224) x (768x224)^T
    gemm_bf16_kernel<<<dim3(CID / 256, 13, BATCH), 256, 0, stream>>>(
        attn, NPAD, (long long)NPOS * NPAD, gT, NPAD, (long long)CID * NPAD,
        NPOS, CID, NPAD, 2,
        ybuf, CID, (long long)NPOS * CID,
        nullptr, nullptr, nullptr, nullptr, nullptr, nullptr);
    // 7) out = BN(y * w_W^T + b_W) + xj : (784x768) x (1536x768)^T fused epilogue
    gemm_bf16_kernel<<<dim3(CH / 256, 13, BATCH), 256, 0, stream>>>(
        ybuf, CID, (long long)NPOS * CID, wTW, CID, 0,
        NPOS, CH, CID, 3,
        d_out, 0, 0,
        b_W, gamma, beta, mean, var, x);
}
